// QRNNLayer_51230369907342
// MI455X (gfx1250) — matrix operand order repaired and checked by reading, not compiled, not run on hardware
//
#include <hip/hip_runtime.h>
#include <math.h>

// ---------------- CDNA5 vector types (trivial, union-safe) ----------------
typedef __attribute__((ext_vector_type(16))) __bf16       v16bf;
typedef __attribute__((ext_vector_type(8)))  float        v8f;
typedef __attribute__((ext_vector_type(4)))  unsigned int u32x4;

#define BATCH   8
#define SEQ     4096
#define OUT     256
#define KDIM    512            // WIN*IN
#define NDIM    768            // 3*OUT
#define BROWS   64             // tokens per block
#define SCHUNK  32             // scan chunk length (2 per block, all 512 threads scan)
#define NSC     (SEQ / SCHUNK) // 128 scan chunks per batch
#define KC      128            // K slab staged in LDS
#define LDA     (KC + 8)       // padded LDS row stride in bf16 (272B rows -> conflict-free b128)
#define NBLK    (SEQ / BROWS)  // 64 blocks per batch

union FragBF { v16bf v; u32x4 u[2]; };

__device__ __forceinline__ float sigmoidf_(float v) {
    return 1.0f / (1.0f + __expf(-v));
}

// ---------------------------------------------------------------------------
// Pass 1: windowed GEMM (bf16 WMMA, f32 accum) + activations + in-chunk scan.
// Block = 512 threads = 16 wave32s.
// Wave w: rg = w&1 -> row tiles {2*rg, 2*rg+1} (32 rows),
//         cgrp = w>>1 -> 6 col-tiles (96 cols). 64x768 per block.
// Each B fragment feeds 2 WMMAs; B loads are ping-ponged one tile ahead so
// WMMA overlaps the LDS pipe instead of draining it (s_wait_dscnt != 0).
// ---------------------------------------------------------------------------
__global__ __launch_bounds__(512)
void qrnn_gemm_scan(const float* __restrict__ x, const float* __restrict__ W,
                    const float* __restrict__ bias,
                    float* __restrict__ ws_c, float* __restrict__ ws_P,
                    float* __restrict__ ws_g,
                    float* __restrict__ wsA, float* __restrict__ wsB) {
    extern __shared__ char smem[];
    __bf16* Wbuf = (__bf16*)smem;                              // NDIM * LDA bf16
    __bf16* Abuf = (__bf16*)(smem + (size_t)NDIM * LDA * 2);   // BROWS * LDA bf16
    float*  ybuf = (float*)smem;                               // BROWS * NDIM f32 (reuses W/A region)

    const int tid  = threadIdx.x;
    const int lane = tid & 31;          // wave32
    const int w    = tid >> 5;          // 0..15
    const int rg   = w & 1;             // row group (2 tiles)
    const int cgrp = w >> 1;            // col group (6 tiles)

    const int blk   = blockIdx.x;
    const int bi    = blk / NBLK;
    const int chunk = blk % NBLK;
    const int s0    = chunk * BROWS;

    v8f acc[2][6] = {};

    const int halfSel = (lane & 16) ? 8 : 0;          // A-matrix K-half select (ISA 7.12.2)
    const int lane15  = lane & 15;
    const int bHalf   = (lane & 16) ? 8 : 0;          // B-matrix K-half select (same as A)

    for (int kc = 0; kc < KDIM / KC; ++kc) {
        // ---- stage W slab: W[n][kc*KC .. +KC) as bf16 into LDS (L2-resident source) ----
        for (int idx = tid; idx < NDIM * KC; idx += 512) {
            const int n  = idx >> 7;          // KC == 128
            const int kk = idx & (KC - 1);
            Wbuf[n * LDA + kk] = (__bf16)W[n * KDIM + kc * KC + kk];
        }
        // ---- stage A slab: windowed input xw[b, s0+r, kg] ----
        for (int idx = tid; idx < BROWS * KC; idx += 512) {
            const int r  = idx >> 7;
            const int kk = idx & (KC - 1);
            const int kg = kc * KC + kk;
            const int s  = s0 + r;
            float v;
            if (kg < 256) {                   // previous-token half of the window
                v = (s == 0) ? 0.0f : x[((bi * SEQ + s - 1) << 8) + kg];
            } else {                          // current-token half
                v = x[((bi * SEQ + s) << 8) + (kg - 256)];
            }
            Abuf[r * LDA + kk] = (__bf16)v;
        }
        // speculative prefetch of next W slab into L2 (global_prefetch_b8)
        if (kc + 1 < KDIM / KC) {
            const float* nw = W + (kc + 1) * KC;
            for (int n = tid; n < NDIM; n += 512)
                __builtin_prefetch(nw + n * KDIM, 0, 0);
        }
        __syncthreads();

        // ---- WMMA over this K slab: 4 k-steps of 32 ----
        #pragma unroll
        for (int ks = 0; ks < KC / 32; ++ks) {
            // A fragments for the wave's two row tiles
            // (lanes 0-15 hold K {0-7,16-23}, lanes 16-31 hold {8-15,24-31})
            const int kbA = ks * 32 + halfSel;
            FragBF a0, a1;
            {
                const __bf16* ap0 = Abuf + (rg * 32 + lane15) * LDA + kbA;
                const __bf16* ap1 = ap0 + 16 * LDA;
                a0.u[0] = *(const u32x4*)(ap0);
                a0.u[1] = *(const u32x4*)(ap0 + 16);
                a1.u[0] = *(const u32x4*)(ap1);
                a1.u[1] = *(const u32x4*)(ap1 + 16);
            }
            // B fragments, ping-ponged one col-tile ahead
            const __bf16* bbase = Wbuf + (cgrp * 96 + lane15) * LDA + ks * 32 + bHalf;
            FragBF bf[2];
            bf[0].u[0] = *(const u32x4*)(bbase);
            bf[0].u[1] = *(const u32x4*)(bbase + 16);
            #pragma unroll
            for (int ct = 0; ct < 6; ++ct) {
                const int cur = ct & 1, nxt = cur ^ 1;
                if (ct < 5) {
                    const __bf16* bp = bbase + (ct + 1) * 16 * LDA;
                    bf[nxt].u[0] = *(const u32x4*)(bp);
                    bf[nxt].u[1] = *(const u32x4*)(bp + 16);
                }
                acc[0][ct] = __builtin_amdgcn_wmma_f32_16x16x32_bf16(
                    false, a0.v, false, bf[cur].v, (short)0, acc[0][ct], false, false);
                acc[1][ct] = __builtin_amdgcn_wmma_f32_16x16x32_bf16(
                    false, a1.v, false, bf[cur].v, (short)0, acc[1][ct], false, false);
            }
        }
        __syncthreads();   // before next slab overwrites LDS
    }

    // ---- spill y tiles to LDS (C/D layout: VGPR g -> row g / g+8 of the tile) ----
    #pragma unroll
    for (int rtl = 0; rtl < 2; ++rtl) {
        #pragma unroll
        for (int ct = 0; ct < 6; ++ct) {
            const int col = cgrp * 96 + ct * 16 + lane15;
            #pragma unroll
            for (int g = 0; g < 8; ++g) {
                const int row = rg * 32 + rtl * 16 + g + halfSel;
                ybuf[row * NDIM + col] = acc[rtl][ct][g];
            }
        }
    }
    __syncthreads();

    // ---- fused activations + fo-pool scan: 512 threads = 256 ch x 2 subchunks ----
    {
        const int ch   = tid & 255;
        const int half = tid >> 8;            // which 32-token subchunk
        const int rb   = half * SCHUNK;
        const float bz = bias[ch];
        const float bfo = bias[OUT + ch];
        const float bo = bias[2 * OUT + ch];
        float c = 0.0f, P = 1.0f;
        for (int r = 0; r < SCHUNK; ++r) {
            const int rr = rb + r;
            const float zv = tanhf(ybuf[rr * NDIM + ch] + bz);
            const float fv = sigmoidf_(ybuf[rr * NDIM + OUT + ch] + bfo);
            const float ov = sigmoidf_(ybuf[rr * NDIM + 2 * OUT + ch] + bo);
            c = fv * c + (1.0f - fv) * zv;    // local scan with c_in = 0
            P *= fv;                          // prefix product of forget gates
            const int gi = ((bi * SEQ + s0 + rr) << 8) + ch;
            ws_c[gi] = c;
            ws_P[gi] = P;
            ws_g[gi] = ov;
        }
        const int ci = (bi * NSC + chunk * 2 + half) * OUT + ch;
        wsA[ci] = P;   // subchunk composite: c_out = A * c_in + B
        wsB[ci] = c;
    }
}

// ---------------------------------------------------------------------------
// Pass 2: sequential combine of the 128 subchunk composites per (batch, channel).
// ---------------------------------------------------------------------------
__global__ __launch_bounds__(256)
void qrnn_chunk_scan(const float* __restrict__ wsA, const float* __restrict__ wsB,
                     float* __restrict__ Cin) {
    const int id = blockIdx.x * blockDim.x + threadIdx.x;   // 2048 total
    if (id >= BATCH * OUT) return;
    const int b  = id >> 8;
    const int ch = id & 255;
    float carry = 0.0f;
    for (int c = 0; c < NSC; ++c) {
        const int idx = (b * NSC + c) * OUT + ch;
        Cin[idx] = carry;                      // exclusive prefix (incoming state)
        carry = wsA[idx] * carry + wsB[idx];
    }
}

// ---------------------------------------------------------------------------
// Pass 3: fixup + output gate, float4 vectorized.
// out = sigmoid(o) * (c_local + P * C_in[subchunk])
// ---------------------------------------------------------------------------
__global__ __launch_bounds__(256)
void qrnn_fixup(const float* __restrict__ ws_c, const float* __restrict__ ws_P,
                const float* __restrict__ ws_g, const float* __restrict__ Cin,
                float* __restrict__ out) {
    const long i = ((long)blockIdx.x * blockDim.x + threadIdx.x) * 4;
    const int  ch = (int)(i & 255);
    const long bs = i >> 8;
    const int  s  = (int)(bs & (SEQ - 1));
    const int  b  = (int)(bs >> 12);
    const int  ci = ((b * NSC + (s >> 5)) << 8) + ch;

    const float4 c = *(const float4*)(ws_c + i);
    const float4 P = *(const float4*)(ws_P + i);
    const float4 g = *(const float4*)(ws_g + i);
    const float4 C = *(const float4*)(Cin + ci);
    float4 o;
    o.x = g.x * (c.x + P.x * C.x);
    o.y = g.y * (c.y + P.y * C.y);
    o.z = g.z * (c.z + P.z * C.z);
    o.w = g.w * (c.w + P.w * C.w);
    *(float4*)(out + i) = o;
}

// ---------------------------------------------------------------------------
extern "C" void kernel_launch(void* const* d_in, const int* in_sizes, int n_in,
                              void* d_out, int out_size, void* d_ws, size_t ws_size,
                              hipStream_t stream) {
    const float* x    = (const float*)d_in[0];   // (8, 4096, 256) f32
    const float* W    = (const float*)d_in[1];   // (768, 512)     f32
    const float* bias = (const float*)d_in[2];   // (768,)         f32
    float* out = (float*)d_out;

    const long TOK = (long)BATCH * SEQ * OUT;    // 8,388,608
    float* ws   = (float*)d_ws;
    float* ws_c = ws;                            // local scan result
    float* ws_P = ws + TOK;                      // prefix forget-product
    float* ws_g = ws + 2 * TOK;                  // sigmoid(o)
    float* wsA  = ws + 3 * TOK;                  // subchunk composite A (Pi f)
    float* wsB  = wsA + (long)BATCH * NSC * OUT;
    float* Cin  = wsB + (long)BATCH * NSC * OUT;

    const size_t shmem = (size_t)NDIM * LDA * 2 + (size_t)BROWS * LDA * 2; // 226,304 B (<320KB WGP LDS)
    hipFuncSetAttribute((const void*)qrnn_gemm_scan,
                        hipFuncAttributeMaxDynamicSharedMemorySize, (int)shmem);

    qrnn_gemm_scan<<<BATCH * NBLK, 512, shmem, stream>>>(x, W, bias,
                                                         ws_c, ws_P, ws_g, wsA, wsB);
    qrnn_chunk_scan<<<(BATCH * OUT + 255) / 256, 256, 0, stream>>>(wsA, wsB, Cin);
    qrnn_fixup<<<(int)(TOK / 4 / 256), 256, 0, stream>>>(ws_c, ws_P, ws_g, Cin, out);
}
